// GraphAutoencoder_19902878450309
// MI455X (gfx1250) — matrix operand order repaired and checked by reading, not compiled, not run on hardware
//
#include <hip/hip_runtime.h>
#include <math.h>

typedef __attribute__((ext_vector_type(16))) _Float16 v16h;
typedef __attribute__((ext_vector_type(8)))  _Float16 v8h;
typedef __attribute__((ext_vector_type(8)))  float    v8f;

#define EPI_NONE    0
#define EPI_RELU    1
#define EPI_RELU_BN 2

#define NB(n) ((int)((((long long)(n)) + 255) / 256))

// ---- order-preserving float<->uint encoding for atomicMax-based segment max ----
__device__ __forceinline__ unsigned encf(float f){
  unsigned u = __float_as_uint(f);
  return (u & 0x80000000u) ? ~u : (u | 0x80000000u);
}
__device__ __forceinline__ float decf(unsigned u){
  return __uint_as_float((u & 0x80000000u) ? (u & 0x7FFFFFFFu) : ~u);
}

// =================== elementwise / scatter kernels ===================

__global__ void k_fill_enc(unsigned* __restrict__ p, int n, float v){
  int i = blockIdx.x*blockDim.x + threadIdx.x;
  if (i < n) p[i] = encf(v);
}

// copy edge_attr into ea[0:E] and scatter-sum into ea[E:E+N] (per-dst), count into cnt
__global__ void k_build_ea(const float* __restrict__ eattr, const int* __restrict__ dstp,
                           float* __restrict__ ea, float* __restrict__ cnt, int E){
  int t = blockIdx.x*blockDim.x + threadIdx.x;
  if (t >= E*16) return;
  int e = t >> 4, j = t & 15;
  float v = eattr[t];
  ea[t] = v;
  int d = dstp[e];
  atomicAdd(&ea[(E + d)*16 + j], v);
  if (j == 0) atomicAdd(&cnt[d], 1.0f);
}

__global__ void k_loop_mean(float* __restrict__ ea, const float* __restrict__ cnt, int E, int Nn){
  int t = blockIdx.x*blockDim.x + threadIdx.x;
  if (t >= Nn*16) return;
  int n = t >> 4;
  ea[(E + n)*16 + (t & 15)] /= fmaxf(cnt[n], 1.0f);
}

// fold We0 (16x256) with ae0 (4x64) -> Ve0 (16x4)
__global__ void k_fold_ve0(const float* __restrict__ We0, const float* __restrict__ ae0,
                           float* __restrict__ Ve0){
  int t = threadIdx.x; if (t >= 64) return;
  int k = t >> 2, h = t & 3;
  float s = 0.f;
  for (int c = 0; c < 64; ++c) s += We0[k*256 + h*64 + c] * ae0[h*64 + c];
  Ve0[t] = s;
}
// fold We1 (16x32) with ae1 (32) -> Ve1 (16)
__global__ void k_fold_ve1(const float* __restrict__ We1, const float* __restrict__ ae1,
                           float* __restrict__ Ve1){
  int k = threadIdx.x; if (k >= 16) return;
  float s = 0.f;
  for (int c = 0; c < 32; ++c) s += We1[k*32 + c] * ae1[c];
  Ve1[k] = s;
}

// a_e0[f,h] = ea[f,:] . Ve0[:,h]
__global__ void k_ae0(const float* __restrict__ ea, const float* __restrict__ Ve0,
                      float* __restrict__ out, int F){
  int t = blockIdx.x*blockDim.x + threadIdx.x;
  if (t >= F*4) return;
  int f = t >> 2, h = t & 3;
  float s = 0.f;
#pragma unroll
  for (int k = 0; k < 16; ++k) s += ea[f*16 + k] * Ve0[k*4 + h];
  out[t] = s;
}
__global__ void k_ae1(const float* __restrict__ ea, const float* __restrict__ Ve1,
                      float* __restrict__ out, int F){
  int f = blockIdx.x*blockDim.x + threadIdx.x;
  if (f >= F) return;
  float s = 0.f;
#pragma unroll
  for (int k = 0; k < 16; ++k) s += ea[f*16 + k] * Ve1[k];
  out[f] = s;
}

// a_s0/a_d0 from xw0 [N,256], heads=4, hid=64
__global__ void k_sd0(const float* __restrict__ xw, const float* __restrict__ as,
                      const float* __restrict__ ad, float* __restrict__ os,
                      float* __restrict__ od, int Nn){
  int t = blockIdx.x*blockDim.x + threadIdx.x;
  if (t >= Nn*4) return;
  int n = t >> 2, h = t & 3;
  float sa = 0.f, sd = 0.f;
  for (int c = 0; c < 64; ++c){
    float v = xw[n*256 + h*64 + c];
    sa += v*as[h*64 + c]; sd += v*ad[h*64 + c];
  }
  os[t] = sa; od[t] = sd;
}
// a_s1/a_d1 from xw1 [N,32], 1 head
__global__ void k_sd1(const float* __restrict__ xw, const float* __restrict__ as,
                      const float* __restrict__ ad, float* __restrict__ os,
                      float* __restrict__ od, int Nn){
  int n = blockIdx.x*blockDim.x + threadIdx.x;
  if (n >= Nn) return;
  float sa = 0.f, sd = 0.f;
  for (int c = 0; c < 32; ++c){
    float v = xw[n*32 + c];
    sa += v*as[c]; sd += v*ad[c];
  }
  os[n] = sa; od[n] = sd;
}

// alpha = leaky_relu(a_s[src]+a_d[dst]+a_e, 0.2); segment max via encoded atomicMax
__global__ void k_alpha_max(const int* __restrict__ src, const int* __restrict__ dstp,
                            const float* __restrict__ asb, const float* __restrict__ adb,
                            float* __restrict__ alpha, unsigned* __restrict__ mbuf,
                            int E, int F, int H){
  int t = blockIdx.x*blockDim.x + threadIdx.x;
  if (t >= F*H) return;
  int f = t / H, h = t - f*H;
  int s = (f < E) ? src[f]  : (f - E);
  int d = (f < E) ? dstp[f] : (f - E);
  float a = asb[s*H + h] + adb[d*H + h] + alpha[t];
  a = (a > 0.f) ? a : 0.2f*a;
  alpha[t] = a;
  atomicMax(&mbuf[d*H + h], encf(a));
}

__global__ void k_exp_sum(const int* __restrict__ dstp, float* __restrict__ alpha,
                          const unsigned* __restrict__ mbuf, float* __restrict__ sbuf,
                          int E, int F, int H){
  int t = blockIdx.x*blockDim.x + threadIdx.x;
  if (t >= F*H) return;
  int f = t / H, h = t - f*H;
  int d = (f < E) ? dstp[f] : (f - E);
  float ex = __expf(alpha[t] - decf(mbuf[d*H + h]));
  alpha[t] = ex;
  atomicAdd(&sbuf[d*H + h], ex);
}

// out[dst, :] += (alpha/sum) * xw[src, :]
__global__ void k_aggregate(const int* __restrict__ src, const int* __restrict__ dstp,
                            const float* __restrict__ alpha, const float* __restrict__ sbuf,
                            const float* __restrict__ xw, float* __restrict__ agg,
                            int E, int F, int H, int HC, int hshift){
  long long t = (long long)blockIdx.x*blockDim.x + threadIdx.x;
  if (t >= (long long)F*HC) return;
  int f = (int)(t / HC);
  int j = (int)(t - (long long)f*HC);
  int h = j >> hshift;
  int s = (f < E) ? src[f]  : (f - E);
  int d = (f < E) ? dstp[f] : (f - E);
  float w = alpha[f*H + h] / (sbuf[d*H + h] + 1e-16f);
  atomicAdd(&agg[(long long)d*HC + j], w * xw[(long long)s*HC + j]);
}

// h = elu(bn(agg + b0))
__global__ void k_bnelu(float* __restrict__ hb, const float* __restrict__ b0,
                        const float* __restrict__ g, const float* __restrict__ bb,
                        float bninv, int total){
  int t = blockIdx.x*blockDim.x + threadIdx.x;
  if (t >= total) return;
  int j = t & 255;
  float v = hb[t] + b0[j];
  v = v*bninv*g[j] + bb[j];
  hb[t] = (v > 0.f) ? v : (__expf(v) - 1.0f);
}

__global__ void k_addbias(float* __restrict__ p, const float* __restrict__ b,
                          int total, int mask){
  int t = blockIdx.x*blockDim.x + threadIdx.x;
  if (t >= total) return;
  p[t] += b[t & mask];
}

// edge_recon = sigmoid(e2 . edW3 + edb3)
__global__ void k_edge_final(const float* __restrict__ e2, const float* __restrict__ w3,
                             const float* __restrict__ b3, float* __restrict__ out,
                             int Mc, int base){
  int r = blockIdx.x*blockDim.x + threadIdx.x;
  if (r >= Mc) return;
  float s = b3[0];
  for (int c = 0; c < 64; ++c) s += e2[r*64 + c]*w3[c];
  out[base + r] = 1.0f/(1.0f + __expf(-s));
}

// transpose + f16-convert + K-pad weights: W[K,N] f32 -> Wt[N,Kpad] f16
__global__ void k_prep_wt(const float* __restrict__ W, _Float16* __restrict__ Wt,
                          int K, int N, int Kpad){
  int t = blockIdx.x*blockDim.x + threadIdx.x;
  if (t >= N*Kpad) return;
  int n = t / Kpad, kp = t - n*Kpad;
  Wt[t] = (_Float16)((kp < K) ? W[kp*N + n] : 0.f);
}

// materialize padded edge features: ef[r, 0:32]=z[src], [32:64]=z[dst], [64:80]=edge_attr, [80:96]=0
__global__ void k_gather_ef(const int* __restrict__ src, const int* __restrict__ dstp,
                            const float* __restrict__ z, const float* __restrict__ eattr,
                            float* __restrict__ ef, int Mc, int base){
  int t = blockIdx.x*blockDim.x + threadIdx.x;
  if (t >= Mc*24) return;           // 24 float4 per row of 96
  int r = t / 24, j4 = t - r*24;
  int e = base + r;
  float4 v;
  if (j4 < 8)       v = ((const float4*)(z + src[e]*32))[j4];
  else if (j4 < 16) v = ((const float4*)(z + dstp[e]*32))[j4 - 8];
  else if (j4 < 20) v = ((const float4*)(eattr + (long long)e*16))[j4 - 16];
  else              v = make_float4(0.f, 0.f, 0.f, 0.f);
  ((float4*)(ef + (long long)r*96))[j4] = v;
}

// =================== WMMA GEMM ===================
// C[M,N] = act(A[M,KK] @ W[KK,N] + bias), KK a compile-time multiple of 32.
// A: f32 row-major, row stride KK.  Wt: f16, transposed [N, KK].
// One wave -> one 16x16 C tile; fully unrolled, branch-free K loop.
template<int EPI, int KK>
__launch_bounds__(128)
__global__ void k_gemm(const float* __restrict__ A, const _Float16* __restrict__ Wt,
                       float* __restrict__ C, int M, int Ncols,
                       const float* __restrict__ bias,
                       const float* __restrict__ bng, const float* __restrict__ bnbb,
                       float bninv){
  int wave = threadIdx.x >> 5;
  int lane = threadIdx.x & 31;
  int tileM = (blockIdx.y*4 + wave)*16;
  int tileN = blockIdx.x*16;
  if (tileM >= M) return;                 // wave-uniform; EXEC all-1 at WMMA
  int r  = lane & 15;
  int hi = lane >> 4;
  const float*    arow = A  + (long long)(tileM + r)*KK;   // A-fragment row
  const _Float16* brow = Wt + (long long)(tileN + r)*KK;   // B-fragment column (transposed)
  int ncol = tileN + r;
  v8f acc = {0.f,0.f,0.f,0.f,0.f,0.f,0.f,0.f};
#pragma unroll
  for (int k0 = 0; k0 < KK; k0 += 32){
    // A 16x32 fragment (ISA 7.12.2): per lane two contiguous 8-float runs.
    float4 a0 = *(const float4*)(arow + k0 + 8*hi);
    float4 a1 = *(const float4*)(arow + k0 + 8*hi + 4);
    float4 a2 = *(const float4*)(arow + k0 + 8*hi + 16);
    float4 a3 = *(const float4*)(arow + k0 + 8*hi + 20);
    v16h a;
    a[0]=(_Float16)a0.x; a[1]=(_Float16)a0.y; a[2]=(_Float16)a0.z; a[3]=(_Float16)a0.w;
    a[4]=(_Float16)a1.x; a[5]=(_Float16)a1.y; a[6]=(_Float16)a1.z; a[7]=(_Float16)a1.w;
    a[8]=(_Float16)a2.x; a[9]=(_Float16)a2.y; a[10]=(_Float16)a2.z; a[11]=(_Float16)a2.w;
    a[12]=(_Float16)a3.x; a[13]=(_Float16)a3.y; a[14]=(_Float16)a3.z; a[15]=(_Float16)a3.w;
    // B 32x16 fragment: same K order as A -> two 8-f16 runs at K=8*hi and K=8*hi+16
    v8h b0 = *(const v8h*)(brow + k0 + 8*hi);
    v8h b1 = *(const v8h*)(brow + k0 + 8*hi + 16);
    v16h b;
#pragma unroll
    for (int i = 0; i < 8; ++i){ b[i] = b0[i]; b[i+8] = b1[i]; }
    acc = __builtin_amdgcn_wmma_f32_16x16x32_f16(false, a, false, b, (short)0, acc,
                                                 false, false);
  }
  float bs = bias ? bias[ncol] : 0.f;
  float gg = (EPI == EPI_RELU_BN) ? bng[ncol]  : 0.f;
  float sh = (EPI == EPI_RELU_BN) ? bnbb[ncol] : 0.f;
#pragma unroll
  for (int j = 0; j < 8; ++j){
    int m = tileM + j + 8*hi;               // C layout: VGPR j -> M=j (+8 for hi lanes)
    float v = acc[j] + bs;
    if (EPI >= EPI_RELU)     v = fmaxf(v, 0.f);
    if (EPI == EPI_RELU_BN)  v = v*bninv*gg + sh;
    C[(long long)m*Ncols + ncol] = v;
  }
}

template<int EPI, int KK>
static void launch_gemm(const float* A, const _Float16* Wt, float* C, int M, int Ncols,
                        const float* bias, const float* g, const float* bb, float bninv,
                        hipStream_t stream){
  dim3 grid((Ncols + 15)/16, (M + 63)/64);
  k_gemm<EPI,KK><<<grid, dim3(128), 0, stream>>>(A, Wt, C, M, Ncols, bias, g, bb, bninv);
}

// =================== host orchestration ===================

extern "C" void kernel_launch(void* const* d_in, const int* in_sizes, int n_in,
                              void* d_out, int out_size, void* d_ws, size_t ws_size,
                              hipStream_t stream){
  const float* x     = (const float*)d_in[0];
  const float* eattr = (const float*)d_in[1];
  const float* W0    = (const float*)d_in[2];
  const float* We0   = (const float*)d_in[3];
  const float* as0   = (const float*)d_in[4];
  const float* ad0   = (const float*)d_in[5];
  const float* ae0   = (const float*)d_in[6];
  const float* b0    = (const float*)d_in[7];
  const float* bn0g  = (const float*)d_in[8];
  const float* bn0b  = (const float*)d_in[9];
  const float* W1    = (const float*)d_in[10];
  const float* We1   = (const float*)d_in[11];
  const float* as1   = (const float*)d_in[12];
  const float* ad1   = (const float*)d_in[13];
  const float* ae1   = (const float*)d_in[14];
  const float* b1    = (const float*)d_in[15];
  const float* ndW1  = (const float*)d_in[16];
  const float* ndb1  = (const float*)d_in[17];
  const float* ndg   = (const float*)d_in[18];
  const float* ndbb  = (const float*)d_in[19];
  const float* ndW2  = (const float*)d_in[20];
  const float* ndb2  = (const float*)d_in[21];
  const float* ndW3  = (const float*)d_in[22];
  const float* ndb3  = (const float*)d_in[23];
  const float* edW1  = (const float*)d_in[24];
  const float* edb1  = (const float*)d_in[25];
  const float* edg   = (const float*)d_in[26];
  const float* edbb  = (const float*)d_in[27];
  const float* edW2  = (const float*)d_in[28];
  const float* edb2  = (const float*)d_in[29];
  const float* edW3  = (const float*)d_in[30];
  const float* edb3  = (const float*)d_in[31];
  const int*   ei    = (const int*)d_in[32];

  const int Nn = in_sizes[0] / 128;   // 50000
  const int E  = in_sizes[1] / 16;    // 800000
  const int F  = E + Nn;              // edges + self loops
  const int* srcp = ei;
  const int* dstp = ei + E;
  const float bninv = 1.0f / sqrtf(1.0f + 1e-5f);

  // ---- workspace layout (f32 region, then f16 region) ----
  float* ws = (float*)d_ws;
  size_t o = 0;
  float* EA     = ws + o; o += (size_t)F*16;
  float* CNT    = ws + o; o += (size_t)Nn;
  float* XW0    = ws + o; o += (size_t)Nn*256;
  float* AS0v   = ws + o; o += (size_t)Nn*4;
  float* AD0v   = ws + o; o += (size_t)Nn*4;
  float* ALPHA0 = ws + o; o += (size_t)F*4;     // a_e0 -> lrelu-alpha -> exp
  float* M0b    = ws + o; o += (size_t)Nn*4;    // encoded uint max
  float* S0b    = ws + o; o += (size_t)Nn*4;
  float* AGG0   = ws + o; o += (size_t)Nn*256;  // aggregation, then h (in place)
  float* XW1    = ws + o; o += (size_t)Nn*32;
  float* AS1v   = ws + o; o += (size_t)Nn;
  float* AD1v   = ws + o; o += (size_t)Nn;
  float* ALPHA1 = ws + o; o += (size_t)F;
  float* M1b    = ws + o; o += (size_t)Nn;
  float* S1b    = ws + o; o += (size_t)Nn;
  float* N1b    = ws + o; o += (size_t)Nn*128;
  float* N2b    = ws + o; o += (size_t)Nn*64;
  float* VE0    = ws + o; o += 64;
  float* VE1    = ws + o; o += 16;
  const int CH = 65536;                         // edge-decoder chunk (multiple of 16)
  float* EFc    = ws + o; o += (size_t)CH*96;   // padded gathered edge features
  float* E1c    = ws + o; o += (size_t)CH*128;
  float* E2c    = ws + o; o += (size_t)CH*64;
  o = (o + 3) & ~(size_t)3;                     // 16B-align the f16 region
  _Float16* hws = (_Float16*)(ws + o);
  size_t oh = 0;
  _Float16* W0t  = hws + oh; oh += 256*128;     // [N=256, Kpad=128]
  _Float16* W1t  = hws + oh; oh += 32*256;      // [32, 256]
  _Float16* nW1t = hws + oh; oh += 128*32;      // [128, 32]
  _Float16* nW2t = hws + oh; oh += 64*128;      // [64, 128]
  _Float16* nW3t = hws + oh; oh += 128*64;      // [128, 64]
  _Float16* eW1t = hws + oh; oh += 128*96;      // [128, 96] (K 80 -> pad 96)
  _Float16* eW2t = hws + oh; oh += 64*128;      // [64, 128]
  if (o*sizeof(float) + oh*sizeof(_Float16) > ws_size) return;

  float* zout = (float*)d_out;                  // [N,32]
  float* nrec = zout + (size_t)Nn*32;           // [N,128]
  float* erec = nrec + (size_t)Nn*128;          // [E]

  // ---- weight prep (transpose + f16 + pad) ----
  k_prep_wt<<<NB(256*128), 256, 0, stream>>>(W0,   W0t, 128, 256, 128);
  k_prep_wt<<<NB(32*256),  256, 0, stream>>>(W1,   W1t, 256, 32,  256);
  k_prep_wt<<<NB(128*32),  256, 0, stream>>>(ndW1, nW1t, 32, 128, 32);
  k_prep_wt<<<NB(64*128),  256, 0, stream>>>(ndW2, nW2t, 128, 64, 128);
  k_prep_wt<<<NB(128*64),  256, 0, stream>>>(ndW3, nW3t, 64, 128, 64);
  k_prep_wt<<<NB(128*96),  256, 0, stream>>>(edW1, eW1t, 80, 128, 96);
  k_prep_wt<<<NB(64*128),  256, 0, stream>>>(edW2, eW2t, 128, 64, 128);

  // ---- init accumulation targets ----
  (void)hipMemsetAsync(EA + (size_t)E*16, 0, sizeof(float)*(size_t)Nn*16, stream);
  (void)hipMemsetAsync(CNT,  0, sizeof(float)*(size_t)Nn,     stream);
  (void)hipMemsetAsync(AGG0, 0, sizeof(float)*(size_t)Nn*256, stream);
  (void)hipMemsetAsync(S0b,  0, sizeof(float)*(size_t)Nn*4,   stream);
  (void)hipMemsetAsync(S1b,  0, sizeof(float)*(size_t)Nn,     stream);
  (void)hipMemsetAsync(zout, 0, sizeof(float)*(size_t)Nn*32,  stream);
  k_fill_enc<<<NB(Nn*4), 256, 0, stream>>>((unsigned*)M0b, Nn*4, -1e30f);
  k_fill_enc<<<NB(Nn),   256, 0, stream>>>((unsigned*)M1b, Nn,   -1e30f);

  // ---- ea = [edge_attr ; scatter-mean self-loop attrs] ----
  k_build_ea<<<NB((long long)E*16), 256, 0, stream>>>(eattr, dstp, EA, CNT, E);
  k_loop_mean<<<NB(Nn*16), 256, 0, stream>>>(EA, CNT, E, Nn);

  // ---- folded edge-attention projections ----
  k_fold_ve0<<<1, 64, 0, stream>>>(We0, ae0, VE0);
  k_fold_ve1<<<1, 16, 0, stream>>>(We1, ae1, VE1);
  k_ae0<<<NB((long long)F*4), 256, 0, stream>>>(EA, VE0, ALPHA0, F);
  k_ae1<<<NB(F), 256, 0, stream>>>(EA, VE1, ALPHA1, F);

  // ---- GAT layer 0 ----
  launch_gemm<EPI_NONE,128>(x, W0t, XW0, Nn, 256, nullptr, nullptr, nullptr, bninv, stream);
  k_sd0<<<NB(Nn*4), 256, 0, stream>>>(XW0, as0, ad0, AS0v, AD0v, Nn);
  k_alpha_max<<<NB((long long)F*4), 256, 0, stream>>>(srcp, dstp, AS0v, AD0v, ALPHA0,
                                                      (unsigned*)M0b, E, F, 4);
  k_exp_sum<<<NB((long long)F*4), 256, 0, stream>>>(dstp, ALPHA0, (unsigned*)M0b, S0b, E, F, 4);
  k_aggregate<<<NB((long long)F*256), 256, 0, stream>>>(srcp, dstp, ALPHA0, S0b, XW0, AGG0,
                                                        E, F, 4, 256, 6);
  k_bnelu<<<NB((long long)Nn*256), 256, 0, stream>>>(AGG0, b0, bn0g, bn0b, bninv, Nn*256);

  // ---- GAT layer 1 ----
  launch_gemm<EPI_NONE,256>(AGG0, W1t, XW1, Nn, 32, nullptr, nullptr, nullptr, bninv, stream);
  k_sd1<<<NB(Nn), 256, 0, stream>>>(XW1, as1, ad1, AS1v, AD1v, Nn);
  k_alpha_max<<<NB(F), 256, 0, stream>>>(srcp, dstp, AS1v, AD1v, ALPHA1,
                                         (unsigned*)M1b, E, F, 1);
  k_exp_sum<<<NB(F), 256, 0, stream>>>(dstp, ALPHA1, (unsigned*)M1b, S1b, E, F, 1);
  k_aggregate<<<NB((long long)F*32), 256, 0, stream>>>(srcp, dstp, ALPHA1, S1b, XW1, zout,
                                                       E, F, 1, 32, 5);
  k_addbias<<<NB(Nn*32), 256, 0, stream>>>(zout, b1, Nn*32, 31);

  // ---- node decoder: z -> 128 (relu,bn) -> 64 (relu) -> 128 ----
  launch_gemm<EPI_RELU_BN,32>(zout, nW1t, N1b, Nn, 128, ndb1, ndg, ndbb, bninv, stream);
  launch_gemm<EPI_RELU,128>(N1b, nW2t, N2b, Nn, 64, ndb2, nullptr, nullptr, bninv, stream);
  launch_gemm<EPI_NONE,64>(N2b, nW3t, nrec, Nn, 128, ndb3, nullptr, nullptr, bninv, stream);

  // ---- edge decoder, chunked: gather [z[src]|z[dst]|edge_attr|0pad] then GEMMs ----
  for (int base = 0; base < E; base += CH){
    int Mc = (E - base < CH) ? (E - base) : CH;   // always a multiple of 16 here
    k_gather_ef<<<NB((long long)Mc*24), 256, 0, stream>>>(srcp, dstp, zout, eattr, EFc, Mc, base);
    launch_gemm<EPI_RELU_BN,96>(EFc, eW1t, E1c, Mc, 128, edb1, edg, edbb, bninv, stream);
    launch_gemm<EPI_RELU,128>(E1c, eW2t, E2c, Mc, 64, edb2, nullptr, nullptr, bninv, stream);
    k_edge_final<<<NB(Mc), 256, 0, stream>>>(E2c, edW3, edb3, erec, Mc, base);
  }
}
